// BiDAF_78847009620106
// MI455X (gfx1250) — compile-verified, matrix operand order checked by reading, not run on hardware
//
#include <hip/hip_runtime.h>
#include <cmath>

typedef __bf16 bf16;
typedef __attribute__((ext_vector_type(16))) __bf16 v16bf;
typedef __attribute__((ext_vector_type(8)))  __bf16 v8bf;
typedef __attribute__((ext_vector_type(8)))  float  v8f;

// Problem constants
#define Bv   32
#define Tv   512
#define Jv   64
#define Ev   300
#define Evp  320    // E padded to multiple of 32 for vectorized WMMA loads
#define Hv   256
#define G4H  1024   // 4*H
#define D2H  512    // 2*H

// ---------------------------------------------------------------------------
// f32 -> bf16 with K-padding:  d[r*Kp + k] = k<K ? s[r*K+k] : 0
// ---------------------------------------------------------------------------
__global__ void convpad_k(const float* __restrict__ s, bf16* __restrict__ d,
                          int rows, int K, int Kp) {
    size_t idx = (size_t)blockIdx.x * blockDim.x + threadIdx.x;
    size_t total = (size_t)rows * Kp;
    size_t stride = (size_t)gridDim.x * blockDim.x;
    for (; idx < total; idx += stride) {
        int r = (int)(idx / Kp);
        int k = (int)(idx % Kp);
        d[idx] = (k < K) ? (bf16)s[(size_t)r * K + k] : (bf16)0.0f;
    }
}

// ---------------------------------------------------------------------------
// Embedding gather -> f32 + bf16, padded row stride Evp (pad cols zeroed)
// ---------------------------------------------------------------------------
__global__ void gather_embed_k(const int* __restrict__ tok, const float* __restrict__ emb,
                               float* __restrict__ xf, bf16* __restrict__ xb, int ntok) {
    size_t idx = (size_t)blockIdx.x * blockDim.x + threadIdx.x;
    size_t total = (size_t)ntok * Evp;
    if (idx >= total) return;
    int row = (int)(idx / Evp);
    int e   = (int)(idx % Evp);
    float v = 0.f;
    if (e < Ev) {
        int t = tok[row];
        v = emb[(size_t)t * Ev + e];
    }
    xf[idx] = v;
    xb[idx] = (bf16)v;
}

// ---------------------------------------------------------------------------
// Highway combine: x = h*t + (1-t)*x  (h,t already ReLU'd, stride Ev; x stride Evp)
// ---------------------------------------------------------------------------
__global__ void highway_combine_k(float* __restrict__ x, bf16* __restrict__ xb,
                                  const float* __restrict__ hb, const float* __restrict__ tb,
                                  int ntok) {
    size_t idx = (size_t)blockIdx.x * blockDim.x + threadIdx.x;
    size_t total = (size_t)ntok * Ev;
    if (idx >= total) return;
    int row = (int)(idx / Ev);
    int col = (int)(idx % Ev);
    size_t xi = (size_t)row * Evp + col;
    float xv = x[xi], hv = hb[idx], tv = tb[idx];
    float nv = hv * tv + (1.f - tv) * xv;
    x[xi]  = nv;
    xb[xi] = (bf16)nv;
}

// ---------------------------------------------------------------------------
// bf16 WMMA GEMM:  C(f32)[M,N] = A[M,K] @ W[N,K]^T + b1 + b2, opt ReLU.
// K, lda, ldw multiples of 32/8 (we pad).  Block 256 thr = 8 waves,
// block tile 64x256, wave tile 32x64 (2x4 WMMA tiles, B-frag reused).
// Full-range tiles are staged with GLOBAL_LOAD_ASYNC_TO_LDS_B128 (ASYNCcnt),
// edge tiles fall back to guarded load + ds_store with zero-fill.
// Optional output row permutation (b*T+t -> t*B+b) via tmT.
// ---------------------------------------------------------------------------
#define LDSP 40  // LDS row pitch in bf16 (80B = multiple of 16B, conflict-free)

__device__ __forceinline__ unsigned lds_addr32(const void* p) {
    // generic pointers into LDS carry the LDS byte offset in the low 32 bits
    return (unsigned)(size_t)p;
}

__device__ __forceinline__ v16bf frag_from_lds(const bf16* base, int rowLocal, int lane) {
    int lm = rowLocal + (lane & 15);
    int kg = (lane >> 4) * 8;
    const v8bf* p0 = reinterpret_cast<const v8bf*>(base + lm * LDSP + kg);
    const v8bf* p1 = reinterpret_cast<const v8bf*>(base + lm * LDSP + 16 + kg);
    v8bf lo = *p0, hi = *p1;
    v16bf r;
#pragma unroll
    for (int i = 0; i < 8; ++i) { r[i] = lo[i]; r[i + 8] = hi[i]; }
    return r;
}

__global__ void gemm_bf16_k(const bf16* __restrict__ A, const bf16* __restrict__ W,
                            const float* __restrict__ b1, const float* __restrict__ b2,
                            float* __restrict__ C, int M, int N, int K,
                            int lda, int ldw, int ldc, int relu, int tmT) {
    __shared__ bf16 As[64 * LDSP];
    __shared__ bf16 Ws[256 * LDSP];
    const int tid  = threadIdx.x;
    const int lane = tid & 31, wid = tid >> 5;
    const int waveM = wid & 1;   // 2 M-groups of 32
    const int waveN = wid >> 1;  // 4 N-groups of 64
    const int blkM = blockIdx.y * 64;
    const int blkN = blockIdx.x * 256;
    const bool fullM = (blkM + 64 <= M);
    const bool fullN = (blkN + 256 <= N);

    v8bf zro;
#pragma unroll
    for (int i = 0; i < 8; ++i) zro[i] = (bf16)0.0f;

    v8f acc[2][4];
#pragma unroll
    for (int i = 0; i < 2; ++i)
#pragma unroll
        for (int t = 0; t < 4; ++t) acc[i][t] = (v8f){0, 0, 0, 0, 0, 0, 0, 0};

    // per-thread tile-load coordinates (8 bf16 = 16B chunks)
    const int am = tid >> 2;          // 0..63
    const int ak = (tid & 3) * 8;     // 0,8,16,24
    const int gmA = blkM + am;
    const unsigned ldsA = lds_addr32(&As[am * LDSP + ak]);

    const int ksteps = K >> 5;
    for (int ks = 0; ks < ksteps; ++ks) {
        const int k0 = ks << 5;
        __syncthreads();
        // ---- A tile: 64x32 = 256 chunks, one per thread ----
        if (fullM) {
            const bf16* p = A + (size_t)gmA * lda + k0 + ak;
            asm volatile("global_load_async_to_lds_b128 %0, %1, off"
                         :: "v"(ldsA), "v"(p) : "memory");
            __builtin_prefetch(p + 32, 0, 3);
        } else {
            v8bf v = zro;
            if (gmA < M) v = *reinterpret_cast<const v8bf*>(A + (size_t)gmA * lda + k0 + ak);
            *reinterpret_cast<v8bf*>(&As[am * LDSP + ak]) = v;
        }
        // ---- W tile: 256x32 = 1024 chunks, four per thread ----
        if (fullN) {
#pragma unroll
            for (int c = 0; c < 4; ++c) {
                int idx = tid + c * 256;
                int n = idx >> 2, k = (idx & 3) * 8;
                const bf16* p = W + (size_t)(blkN + n) * ldw + k0 + k;
                unsigned ldsW = lds_addr32(&Ws[n * LDSP + k]);
                asm volatile("global_load_async_to_lds_b128 %0, %1, off"
                             :: "v"(ldsW), "v"(p) : "memory");
            }
        } else {
#pragma unroll
            for (int c = 0; c < 4; ++c) {
                int idx = tid + c * 256;
                int n = idx >> 2, k = (idx & 3) * 8;
                int gn = blkN + n;
                v8bf v = zro;
                if (gn < N) v = *reinterpret_cast<const v8bf*>(W + (size_t)gn * ldw + k0 + k);
                *reinterpret_cast<v8bf*>(&Ws[n * LDSP + k]) = v;
            }
        }
        asm volatile("s_wait_asynccnt 0" ::: "memory");
        __syncthreads();

        v16bf afr[2];
#pragma unroll
        for (int i = 0; i < 2; ++i)
            afr[i] = frag_from_lds(As, (waveM * 2 + i) * 16, lane);
#pragma unroll
        for (int t = 0; t < 4; ++t) {
            v16bf bfr = frag_from_lds(Ws, (waveN * 4 + t) * 16, lane);
#pragma unroll
            for (int i = 0; i < 2; ++i)
                acc[i][t] = __builtin_amdgcn_wmma_f32_16x16x32_bf16(
                    false, afr[i], false, bfr, (short)0, acc[i][t], false, false);
        }
    }

    // epilogue
#pragma unroll
    for (int t = 0; t < 4; ++t) {
        int col = blkN + waveN * 64 + t * 16 + (lane & 15);
        float bias = 0.f;
        if (col < N) {
            if (b1) bias += b1[col];
            if (b2) bias += b2[col];
        }
#pragma unroll
        for (int i = 0; i < 2; ++i) {
#pragma unroll
            for (int r = 0; r < 8; ++r) {
                int row = blkM + (waveM * 2 + i) * 16 + ((lane < 16) ? r : r + 8);
                if (row < M && col < N) {
                    float v = acc[i][t][r] + bias;
                    if (relu) v = fmaxf(v, 0.f);
                    int orow = tmT ? ((row % tmT) * Bv + row / tmT) : row;
                    C[(size_t)orow * ldc + col] = v;
                }
            }
        }
    }
}

// ---------------------------------------------------------------------------
// Persistent LSTM recurrence, one workgroup per direction (grid = 2).
// 1024 threads = 32 waves. h (bf16) in LDS, c in registers. Xg is TIME-MAJOR
// ([t][b][4H]) so each step streams one contiguous 128KB block.
// ---------------------------------------------------------------------------
#define HSP 264  // LDS pitch for h (528B, 16B aligned, conflict-free)

__global__ void lstm_dir_k(const float* __restrict__ Xg, size_t dirStride,
                           const bf16* __restrict__ Whh, float* __restrict__ outF,
                           bf16* __restrict__ outB, int T, int ldo) {
    __shared__ bf16 hS[32 * HSP];
    const int dir  = blockIdx.x;
    const int tid  = threadIdx.x;
    const int lane = tid & 31, wid = tid >> 5;
    const int rowBase = (wid >> 4) * 16;  // 0 or 16
    const int nG      = wid & 15;         // 0..15
    const float* Xd = Xg + (size_t)dir * dirStride;
    const bf16*  Wd = Whh + (size_t)dir * G4H * Hv;
    const int dirOff = dir * Hv;
    const int rsel = (lane < 16) ? 0 : 8;

    for (int i = tid; i < 32 * HSP; i += 1024) hS[i] = (bf16)0.0f;
    float cReg[8];
#pragma unroll
    for (int r = 0; r < 8; ++r) cReg[r] = 0.f;
    __syncthreads();

    for (int t = 0; t < T; ++t) {
        const int tt = dir ? (T - 1 - t) : t;
        const float* Xt = Xd + (size_t)tt * Bv * G4H;   // time-major block
        v8f acc[4];
#pragma unroll
        for (int g = 0; g < 4; ++g) {
            const int col = g * Hv + nG * 16 + (lane & 15);
#pragma unroll
            for (int r = 0; r < 8; ++r) {
                int row = rowBase + rsel + r;  // batch index
                acc[g][r] = Xt[(size_t)row * G4H + col];
            }
        }
#pragma unroll
        for (int ks = 0; ks < 8; ++ks) {
            const int kk = ks * 32;
            int lm = rowBase + (lane & 15);
            int kg = (lane >> 4) * 8;
            const v8bf* pa0 = reinterpret_cast<const v8bf*>(&hS[lm * HSP + kk + kg]);
            const v8bf* pa1 = reinterpret_cast<const v8bf*>(&hS[lm * HSP + kk + 16 + kg]);
            v8bf alo = *pa0, ahi = *pa1;
            v16bf af;
#pragma unroll
            for (int i = 0; i < 8; ++i) { af[i] = alo[i]; af[i + 8] = ahi[i]; }
#pragma unroll
            for (int g = 0; g < 4; ++g) {
                int nrow = g * Hv + nG * 16 + (lane & 15);
                const v8bf* pb0 = reinterpret_cast<const v8bf*>(Wd + (size_t)nrow * Hv + kk + kg);
                const v8bf* pb1 = reinterpret_cast<const v8bf*>(Wd + (size_t)nrow * Hv + kk + 16 + kg);
                v8bf blo = *pb0, bhi = *pb1;
                v16bf bfr;
#pragma unroll
                for (int i = 0; i < 8; ++i) { bfr[i] = blo[i]; bfr[i + 8] = bhi[i]; }
                acc[g] = __builtin_amdgcn_wmma_f32_16x16x32_bf16(
                    false, af, false, bfr, (short)0, acc[g], false, false);
            }
        }
        __syncthreads();  // all reads of h(t-1) done before overwrite
        const int n = nG * 16 + (lane & 15);
#pragma unroll
        for (int r = 0; r < 8; ++r) {
            int row = rowBase + rsel + r;
            float iv = acc[0][r], fv = acc[1][r], gv = acc[2][r], ov = acc[3][r];
            float si = 1.f / (1.f + __expf(-iv));
            float sf = 1.f / (1.f + __expf(-fv));
            float so = 1.f / (1.f + __expf(-ov));
            float cn = sf * cReg[r] + si * tanhf(gv);
            cReg[r] = cn;
            float hv = so * tanhf(cn);
            size_t oidx = ((size_t)row * T + tt) * ldo + dirOff + n;
            outF[oidx] = hv;
            if (outB) outB[oidx] = (bf16)hv;
            hS[row * HSP + n] = (bf16)hv;
        }
        __syncthreads();  // h(t) visible to all waves
    }
}

// ---------------------------------------------------------------------------
// Attention helpers
// ---------------------------------------------------------------------------
__global__ void rowdot512_k(const float* __restrict__ X, const float* __restrict__ w,
                            float* __restrict__ out, int nrows) {
    int wid = threadIdx.x >> 5, lane = threadIdx.x & 31;
    int row = blockIdx.x * 8 + wid;
    if (row >= nrows) return;
    const float* xr = X + (size_t)row * D2H;
    float acc = 0.f;
    for (int i = lane; i < D2H; i += 32) acc = fmaf(xr[i], w[i], acc);
    for (int s = 16; s > 0; s >>= 1) acc += __shfl_down(acc, s, 32);
    if (lane == 0) out[row] = acc;
}

// One block (64 threads) per (b,t): trilinear scores over j, softmax_j, c2q.
__global__ void att_scores_k(const float* __restrict__ Hc, const float* __restrict__ U,
                             const float* __restrict__ w3, const float* __restrict__ s1,
                             const float* __restrict__ s2, const float* __restrict__ attb,
                             float* __restrict__ c2q, float* __restrict__ rmax) {
    const int bt = blockIdx.x;
    const int b  = bt >> 9;  // T = 512
    const int j  = threadIdx.x;
    __shared__ float hw[D2H];
    __shared__ float Srow[Jv];
    __shared__ float red[Jv];
    const float* hcRow = Hc + (size_t)bt * D2H;
    for (int d = j; d < D2H; d += Jv) hw[d] = hcRow[d] * w3[d];
    __syncthreads();
    const float* uRow = U + ((size_t)b * Jv + j) * D2H;
    float acc = 0.f;
    for (int d = 0; d < D2H; ++d) acc = fmaf(hw[d], uRow[d], acc);
    float sv = acc + s1[bt] + s2[b * Jv + j] + attb[0];
    red[j] = sv;
    __syncthreads();
    for (int s = 32; s > 0; s >>= 1) { if (j < s) red[j] = fmaxf(red[j], red[j + s]); __syncthreads(); }
    float mx = red[0];
    __syncthreads();
    float e = __expf(sv - mx);
    red[j] = e;
    __syncthreads();
    for (int s = 32; s > 0; s >>= 1) { if (j < s) red[j] += red[j + s]; __syncthreads(); }
    float inv = 1.f / red[0];
    __syncthreads();
    Srow[j] = e * inv;
    if (j == 0) rmax[bt] = mx;
    __syncthreads();
    for (int d = j; d < D2H; d += Jv) {
        float a2 = 0.f;
        for (int jj = 0; jj < Jv; ++jj)
            a2 = fmaf(Srow[jj], U[((size_t)b * Jv + jj) * D2H + d], a2);
        c2q[(size_t)bt * D2H + d] = a2;
    }
}

// One block (256 threads) per batch: softmax_t over rowmax, then q2c[b,d].
__global__ void q2c_k(const float* __restrict__ rmax, const float* __restrict__ Hc,
                      float* __restrict__ q2c) {
    const int b = blockIdx.x;
    const int tid = threadIdx.x;
    __shared__ float pr[Tv];
    __shared__ float red[256];
    float v0 = rmax[b * Tv + tid];
    float v1 = rmax[b * Tv + tid + 256];
    red[tid] = fmaxf(v0, v1);
    __syncthreads();
    for (int s = 128; s > 0; s >>= 1) { if (tid < s) red[tid] = fmaxf(red[tid], red[tid + s]); __syncthreads(); }
    float mx = red[0];
    __syncthreads();
    float e0 = __expf(v0 - mx), e1 = __expf(v1 - mx);
    pr[tid] = e0; pr[tid + 256] = e1;
    red[tid] = e0 + e1;
    __syncthreads();
    for (int s = 128; s > 0; s >>= 1) { if (tid < s) red[tid] += red[tid + s]; __syncthreads(); }
    float inv = 1.f / red[0];
    __syncthreads();
    pr[tid] *= inv; pr[tid + 256] *= inv;
    __syncthreads();
    for (int d = tid; d < D2H; d += 256) {
        float acc = 0.f;
        for (int t = 0; t < Tv; ++t)
            acc = fmaf(pr[t], Hc[((size_t)b * Tv + t) * D2H + d], acc);
        q2c[(size_t)b * D2H + d] = acc;
    }
}

// G = [Hc, c2q, Hc*c2q, Hc*q2c]  -> f32 and bf16
__global__ void build_G_k(const float* __restrict__ Hc, const float* __restrict__ c2q,
                          const float* __restrict__ q2c, float* __restrict__ G,
                          bf16* __restrict__ Gb, int total) {
    int idx = blockIdx.x * blockDim.x + threadIdx.x;
    if (idx >= total) return;
    int d  = idx & (D2H - 1);
    int bt = idx >> 9;
    int b  = bt >> 9;
    float hc = Hc[idx], cq = c2q[idx], qc = q2c[(size_t)b * D2H + d];
    size_t g0 = (size_t)bt * (4 * D2H) + d;
    float v0 = hc, v1 = cq, v2 = hc * cq, v3 = hc * qc;
    G[g0]            = v0;
    G[g0 + D2H]      = v1;
    G[g0 + 2 * D2H]  = v2;
    G[g0 + 3 * D2H]  = v3;
    Gb[g0]           = (bf16)v0;
    Gb[g0 + D2H]     = (bf16)v1;
    Gb[g0 + 2 * D2H] = (bf16)v2;
    Gb[g0 + 3 * D2H] = (bf16)v3;
}

// start/end pointer logits: out[bt] = concat(G[bt](2048), M[bt](512)) . pw + pb
__global__ void out_dot_k(const float* __restrict__ G, const float* __restrict__ M,
                          const float* __restrict__ pw, const float* __restrict__ pb,
                          float* __restrict__ out, int nrows) {
    int wid = threadIdx.x >> 5, lane = threadIdx.x & 31;
    int row = blockIdx.x * 8 + wid;
    if (row >= nrows) return;
    const float* g = G + (size_t)row * 2048;
    const float* m = M + (size_t)row * D2H;
    float acc = 0.f;
    for (int i = lane; i < 2048; i += 32) acc = fmaf(g[i], pw[i], acc);
    for (int i = lane; i < D2H; i += 32) acc = fmaf(m[i], pw[2048 + i], acc);
    for (int s = 16; s > 0; s >>= 1) acc += __shfl_down(acc, s, 32);
    if (lane == 0) out[row] = acc + pb[0];
}

// ---------------------------------------------------------------------------
// Host orchestration
// ---------------------------------------------------------------------------
static inline void gemm(const bf16* A, const bf16* W, const float* b1, const float* b2,
                        float* C, int M, int N, int K, int lda, int ldw, int ldc,
                        int relu, int tmT, hipStream_t s) {
    dim3 g((N + 255) / 256, (M + 63) / 64);
    gemm_bf16_k<<<g, 256, 0, s>>>(A, W, b1, b2, C, M, N, K, lda, ldw, ldc, relu, tmT);
}

extern "C" void kernel_launch(void* const* d_in, const int* in_sizes, int n_in,
                              void* d_out, int out_size, void* d_ws, size_t ws_size,
                              hipStream_t stream) {
    (void)in_sizes; (void)n_in; (void)out_size; (void)ws_size;
    const int*   q_tok    = (const int*)d_in[0];
    const int*   c_tok    = (const int*)d_in[1];
    const float* emb      = (const float*)d_in[2];
    const float* hw_lin_w = (const float*)d_in[3];
    const float* hw_lin_b = (const float*)d_in[4];
    const float* hw_gat_w = (const float*)d_in[5];
    const float* hw_gat_b = (const float*)d_in[6];
    const float* ctx_wih  = (const float*)d_in[7];
    const float* ctx_whh  = (const float*)d_in[8];
    const float* ctx_bih  = (const float*)d_in[9];
    const float* ctx_bhh  = (const float*)d_in[10];
    const float* m1_wih   = (const float*)d_in[11];
    const float* m1_whh   = (const float*)d_in[12];
    const float* m1_bih   = (const float*)d_in[13];
    const float* m1_bhh   = (const float*)d_in[14];
    const float* m2_wih   = (const float*)d_in[15];
    const float* m2_whh   = (const float*)d_in[16];
    const float* m2_bih   = (const float*)d_in[17];
    const float* m2_bhh   = (const float*)d_in[18];
    const float* dc_wih   = (const float*)d_in[19];
    const float* dc_whh   = (const float*)d_in[20];
    const float* dc_bih   = (const float*)d_in[21];
    const float* dc_bhh   = (const float*)d_in[22];
    const float* att_w    = (const float*)d_in[23];
    const float* att_b    = (const float*)d_in[24];
    const float* p1_w     = (const float*)d_in[25];
    const float* p1_b     = (const float*)d_in[26];
    const float* p2_w     = (const float*)d_in[27];
    const float* p2_b     = (const float*)d_in[28];
    float* out = (float*)d_out;

    // bump allocator on workspace
    char* base = (char*)d_ws;
    size_t off = 0;
    auto alloc = [&](size_t bytes) -> void* {
        off = (off + 255) & ~(size_t)255;
        void* p = base + off;
        off += bytes;
        return p;
    };
    auto convW = [&](const float* src, int rows, int K, int Kp) -> bf16* {
        size_t n = (size_t)rows * Kp;
        bf16* d = (bf16*)alloc(n * sizeof(bf16));
        convpad_k<<<(unsigned)((n + 1023) / 1024), 256, 0, stream>>>(src, d, rows, K, Kp);
        return d;
    };

    const int NQ = Bv * Jv;   // 2048 question tokens
    const int NC = Bv * Tv;   // 16384 context tokens

    // bf16 weight copies (K padded where needed)
    bf16* w_hwl = convW(hw_lin_w, 2 * Ev, Ev, Evp);
    bf16* w_hwg = convW(hw_gat_w, 2 * Ev, Ev, Evp);
    bf16* w_cih = convW(ctx_wih, 2 * G4H, Ev, Evp);
    bf16* w_chh = convW(ctx_whh, 2 * G4H, Hv, Hv);
    bf16* w_1ih = convW(m1_wih, 2 * G4H, 2048, 2048);
    bf16* w_1hh = convW(m1_whh, 2 * G4H, Hv, Hv);
    bf16* w_2ih = convW(m2_wih, 2 * G4H, D2H, D2H);
    bf16* w_2hh = convW(m2_whh, 2 * G4H, Hv, Hv);
    bf16* w_dih = convW(dc_wih, 2 * G4H, D2H, D2H);
    bf16* w_dhh = convW(dc_whh, 2 * G4H, Hv, Hv);

    // activations / intermediates (x buffers use padded stride Evp)
    float* xq   = (float*)alloc((size_t)NQ * Evp * 4);
    bf16*  xqb  = (bf16*) alloc((size_t)NQ * Evp * 2);
    float* xc   = (float*)alloc((size_t)NC * Evp * 4);
    bf16*  xcb  = (bf16*) alloc((size_t)NC * Evp * 2);
    float* tmpH = (float*)alloc((size_t)NC * Ev * 4);
    float* tmpT = (float*)alloc((size_t)NC * Ev * 4);
    float* qrep = (float*)alloc((size_t)NQ * D2H * 4);
    float* crep = (float*)alloc((size_t)NC * D2H * 4);
    float* Xg   = (float*)alloc((size_t)2 * NC * G4H * 4);   // reused by every LSTM
    float* s1   = (float*)alloc((size_t)NC * 4);
    float* s2   = (float*)alloc((size_t)NQ * 4);
    float* rmax = (float*)alloc((size_t)NC * 4);
    float* c2q  = (float*)alloc((size_t)NC * D2H * 4);
    float* q2c  = (float*)alloc((size_t)Bv * D2H * 4);
    float* Gf   = (float*)alloc((size_t)NC * 2048 * 4);
    bf16*  Gb   = (bf16*) alloc((size_t)NC * 2048 * 2);
    float* M1   = (float*)alloc((size_t)NC * D2H * 4);
    bf16*  M1b  = (bf16*) alloc((size_t)NC * D2H * 2);
    float* Mf   = (float*)alloc((size_t)NC * D2H * 4);
    bf16*  Mb   = (bf16*) alloc((size_t)NC * D2H * 2);
    float* M2   = (float*)alloc((size_t)NC * D2H * 4);

    // 1) embedding gather
    gather_embed_k<<<(NQ * Evp + 255) / 256, 256, 0, stream>>>(q_tok, emb, xq, xqb, NQ);
    gather_embed_k<<<(NC * Evp + 255) / 256, 256, 0, stream>>>(c_tok, emb, xc, xcb, NC);

    // 2) highway (2 layers), question and context
    for (int layer = 0; layer < 2; ++layer) {
        const bf16* lw = w_hwl + (size_t)layer * Ev * Evp;
        const bf16* gw = w_hwg + (size_t)layer * Ev * Evp;
        const float* lb = hw_lin_b + layer * Ev;
        const float* gb = hw_gat_b + layer * Ev;
        gemm(xqb, lw, lb, nullptr, tmpH, NQ, Ev, Evp, Evp, Evp, Ev, 1, 0, stream);
        gemm(xqb, gw, gb, nullptr, tmpT, NQ, Ev, Evp, Evp, Evp, Ev, 1, 0, stream);
        highway_combine_k<<<(NQ * Ev + 255) / 256, 256, 0, stream>>>(xq, xqb, tmpH, tmpT, NQ);
        gemm(xcb, lw, lb, nullptr, tmpH, NC, Ev, Evp, Evp, Evp, Ev, 1, 0, stream);
        gemm(xcb, gw, gb, nullptr, tmpT, NC, Ev, Evp, Evp, Evp, Ev, 1, 0, stream);
        highway_combine_k<<<(NC * Ev + 255) / 256, 256, 0, stream>>>(xc, xcb, tmpH, tmpT, NC);
    }

    // 3) context-encoder BiLSTM: question then context (Xg written time-major)
    for (int dir = 0; dir < 2; ++dir)
        gemm(xqb, w_cih + (size_t)dir * G4H * Evp, ctx_bih + dir * G4H, ctx_bhh + dir * G4H,
             Xg + (size_t)dir * NQ * G4H, NQ, G4H, Evp, Evp, Evp, G4H, 0, Jv, stream);
    lstm_dir_k<<<2, 1024, 0, stream>>>(Xg, (size_t)NQ * G4H, w_chh, qrep, nullptr, Jv, D2H);

    for (int dir = 0; dir < 2; ++dir)
        gemm(xcb, w_cih + (size_t)dir * G4H * Evp, ctx_bih + dir * G4H, ctx_bhh + dir * G4H,
             Xg + (size_t)dir * NC * G4H, NC, G4H, Evp, Evp, Evp, G4H, 0, Tv, stream);
    lstm_dir_k<<<2, 1024, 0, stream>>>(Xg, (size_t)NC * G4H, w_chh, crep, nullptr, Tv, D2H);

    // 4) BiDAF attention
    rowdot512_k<<<(NC + 7) / 8, 256, 0, stream>>>(crep, att_w, s1, NC);          // Hc . w1
    rowdot512_k<<<(NQ + 7) / 8, 256, 0, stream>>>(qrep, att_w + D2H, s2, NQ);    // U . w2
    att_scores_k<<<NC, Jv, 0, stream>>>(crep, qrep, att_w + 2 * D2H, s1, s2, att_b, c2q, rmax);
    q2c_k<<<Bv, 256, 0, stream>>>(rmax, crep, q2c);
    build_G_k<<<(NC * D2H + 255) / 256, 256, 0, stream>>>(crep, c2q, q2c, Gf, Gb, NC * D2H);

    // 5) modeling layer 1: BiLSTM over G (K=2048)
    for (int dir = 0; dir < 2; ++dir)
        gemm(Gb, w_1ih + (size_t)dir * G4H * 2048, m1_bih + dir * G4H, m1_bhh + dir * G4H,
             Xg + (size_t)dir * NC * G4H, NC, G4H, 2048, 2048, 2048, G4H, 0, Tv, stream);
    lstm_dir_k<<<2, 1024, 0, stream>>>(Xg, (size_t)NC * G4H, w_1hh, M1, M1b, Tv, D2H);

    // 6) modeling layer 2: BiLSTM over M1 (K=512)
    for (int dir = 0; dir < 2; ++dir)
        gemm(M1b, w_2ih + (size_t)dir * G4H * D2H, m2_bih + dir * G4H, m2_bhh + dir * G4H,
             Xg + (size_t)dir * NC * G4H, NC, G4H, D2H, D2H, D2H, G4H, 0, Tv, stream);
    lstm_dir_k<<<2, 1024, 0, stream>>>(Xg, (size_t)NC * G4H, w_2hh, Mf, Mb, Tv, D2H);

    // 7) start logits
    out_dot_k<<<(NC + 7) / 8, 256, 0, stream>>>(Gf, Mf, p1_w, p1_b, out, NC);

    // 8) decoder BiLSTM over M (K=512)
    for (int dir = 0; dir < 2; ++dir)
        gemm(Mb, w_dih + (size_t)dir * G4H * D2H, dc_bih + dir * G4H, dc_bhh + dir * G4H,
             Xg + (size_t)dir * NC * G4H, NC, G4H, D2H, D2H, D2H, G4H, 0, Tv, stream);
    lstm_dir_k<<<2, 1024, 0, stream>>>(Xg, (size_t)NC * G4H, w_dhh, M2, nullptr, Tv, D2H);

    // 9) end logits
    out_dot_k<<<(NC + 7) / 8, 256, 0, stream>>>(Gf, M2, p2_w, p2_b, out + NC, NC);
}
